// LSTM_27341761806379
// MI455X (gfx1250) — matrix operand order repaired and checked by reading, not compiled, not run on hardware
//
#include <hip/hip_runtime.h>

typedef __bf16 bf16_t;
typedef __attribute__((ext_vector_type(16))) __bf16 v16bf;
typedef __attribute__((ext_vector_type(8)))  __bf16 v8bf;
typedef __attribute__((ext_vector_type(8)))  float  v8f;

constexpr int T_STEPS = 1024;
constexpr int H       = 512;
constexpr int B       = 64;
constexpr int TILE    = 16;
constexpr int NT_M    = H / TILE;          // 32 row tiles = barrier group size
constexpr int NGRP    = NT_M * (B / TILE); // 128 workgroups
constexpr int KT      = H / 32;            // 16 k-tiles per GEMM
constexpr int LDSK    = H + 8;             // padded K stride (520 elems / 1040 B row)
constexpr int HB      = H * B;
constexpr int SXB     = TILE * LDSK;       // LDS elems per x slice buffer

__device__ __forceinline__ void group_barrier(unsigned* bar, unsigned target) {
  __threadfence();
  __syncthreads();
  if (threadIdx.x == 0) {
    __hip_atomic_fetch_add(bar, 1u, __ATOMIC_RELEASE, __HIP_MEMORY_SCOPE_AGENT);
    while (__hip_atomic_load(bar, __ATOMIC_ACQUIRE, __HIP_MEMORY_SCOPE_AGENT) < target) {
      __builtin_amdgcn_s_sleep(2);
    }
  }
  __syncthreads();
  __threadfence();
}

// CDNA5 async global->LDS copy (16B/lane, ASYNCcnt-tracked)
__device__ __forceinline__ void async_copy_b128(unsigned ldsoff, unsigned goff,
                                                const bf16_t* sbase) {
  asm volatile("global_load_async_to_lds_b128 %0, %1, %2"
               :: "v"(ldsoff), "v"(goff), "s"(sbase) : "memory");
}
__device__ __forceinline__ void wait_async4() {
  asm volatile("s_wait_asynccnt 4" ::: "memory");
}
__device__ __forceinline__ void wait_async0() {
  asm volatile("s_wait_asynccnt 0" ::: "memory");
}

#define BSHUF(lo, hi) __builtin_shufflevector(lo, hi, 0,1,2,3,4,5,6,7,8,9,10,11,12,13,14,15)

// ---- manually sequenced B-fragment loads (invisible to the waitcnt pass) ----
__device__ __forceinline__ void ld_lds_pair(v8bf& lo, v8bf& hi, unsigned addr, int off) {
  asm volatile("ds_load_b128 %0, %2 offset:%3\n\t"
               "ds_load_b128 %1, %2 offset:%4"
               : "=v"(lo), "=v"(hi)
               : "v"(addr), "i"(off), "i"(off + 16));
}
__device__ __forceinline__ void ld_glb_pair(v8bf& lo, v8bf& hi, unsigned voff,
                                            const bf16_t* sbase, int off) {
  asm volatile("global_load_b128 %0, %2, %3 offset:%4\n\t"
               "global_load_b128 %1, %2, %3 offset:%5"
               : "=v"(lo), "=v"(hi)
               : "v"(voff), "s"(sbase), "i"(off), "i"(off + 16));
}
// Manual counter waits; "+v"(b) forges a data dep so the consuming WMMA stays below.
#define MAKE_WAIT(NAME, STR) \
  __device__ __forceinline__ void NAME(v16bf& b) { asm volatile(STR : "+v"(b) :: "memory"); }
MAKE_WAIT(wait_ds4, "s_wait_dscnt 0x4")
MAKE_WAIT(wait_ds2, "s_wait_dscnt 0x2")
MAKE_WAIT(wait_ds0, "s_wait_dscnt 0x0")
MAKE_WAIT(wait_ld4, "s_wait_loadcnt 0x4")
MAKE_WAIT(wait_ld2, "s_wait_loadcnt 0x2")
MAKE_WAIT(wait_ld0, "s_wait_loadcnt 0x0")

__global__ __launch_bounds__(256)
void lstm_persistent_wmma(const float* __restrict__ in_seq,
                          const float* __restrict__ Wxi, const float* __restrict__ Wxf,
                          const float* __restrict__ Wxo, const float* __restrict__ Wxg,
                          const float* __restrict__ Whi, const float* __restrict__ Whf,
                          const float* __restrict__ Who, const float* __restrict__ Whg,
                          float* __restrict__ out,
                          unsigned short* __restrict__ xT_raw,   // bf16 [3][B][H] x ring (transposed)
                          unsigned short* __restrict__ hT_raw,   // bf16 [2][B][H] h (transposed)
                          unsigned* __restrict__ bar)
{
  bf16_t* xT = reinterpret_cast<bf16_t*>(xT_raw);
  bf16_t* hT = reinterpret_cast<bf16_t*>(hT_raw);

  __shared__ __attribute__((aligned(16))) bf16_t sx[2][SXB];   // x slice ping-pong
  __shared__ float gacc[8][TILE * TILE];

  const int tid  = threadIdx.x;
  const int wv   = tid >> 5;
  const int lane = tid & 31;
  const int m0   = (blockIdx.x >> 2) * TILE;
  const int n0   = (blockIdx.x & 3) * TILE;
  unsigned* gbar = bar + (blockIdx.x & 3) * 16;

  // gate order: 0=i,1=f,2=g(cell),3=o ; part 0 -> Wx@x (LDS), part 1 -> Wh@h (global)
  const int gate = wv >> 1;
  const int part = wv & 1;
  const float* WxA[4] = {Wxi, Wxf, Wxg, Wxo};
  const float* WhA[4] = {Whi, Whf, Whg, Who};
  const float* W = part ? WhA[gate] : WxA[gate];

  // ---- weights -> registers as WMMA A fragments (held for all 1024 steps) ----
  const int arow  = m0 + (lane & 15);
  const int akoff = (lane < 16) ? 0 : 16;
  v16bf afrag[KT];
#pragma unroll
  for (int kt = 0; kt < KT; ++kt) {
    const float* s0 = W + (size_t)arow * H + kt * 32 + akoff;
    const float* s1 = s0 + 8;
    v16bf a;
#pragma unroll
    for (int j = 0; j < 8; ++j) { a[j] = (bf16_t)s0[j]; a[8 + j] = (bf16_t)s1[j]; }
    afrag[kt] = a;
  }

  const int em = tid >> 4, en = tid & 15;
  const unsigned sx_base = (unsigned)(unsigned long long)(const void*)&sx[0][0];
  const size_t   tcol    = (size_t)(n0 + en) * H + (m0 + em);   // transposed scatter index

  // per-lane B-fragment byte offsets
  const unsigned bvoff = (unsigned)(((lane & 15) * H + ((lane < 16) ? 0 : 16)) * 2);
  const unsigned lvoff = (unsigned)(((lane & 15) * LDSK + ((lane < 16) ? 0 : 16)) * 2);

  // ---- bootstrap: transpose x_0 -> ring0, x_1 -> ring1 ; hT[0] tile = 0 ----
  {
    const size_t oi = (size_t)(m0 + em) * B + n0 + en;
    xT[tcol]      = (bf16_t)in_seq[oi];
    xT[HB + tcol] = (bf16_t)in_seq[HB + oi];
    hT[tcol]      = (bf16_t)0.0f;
  }
  unsigned bt = NT_M;
  group_barrier(gbar, bt);

  // issue async fill of sx[0] with x_0 (completion enforced inside the loop)
  {
    const bf16_t* xsrc = xT + (size_t)n0 * H;
#pragma unroll
    for (int i = 0; i < 4; ++i) {
      unsigned g = (unsigned)(i * 256 + tid) * 16;
      unsigned l = (g >> 10) * (LDSK * 2) + (g & 1023);
      async_copy_b128(sx_base + l, g, xsrc);
    }
  }

  float c_reg = 0.0f;

#pragma unroll 1
  for (int t = 0; t < T_STEPS; ++t) {
    // ---- fire-and-forget: async fill sx[(t+1)&1] with x_{t+1} ----
    if (t + 1 < T_STEPS) {
      const bf16_t* xsrc = xT + (size_t)((t + 1) % 3) * HB + (size_t)n0 * H;
      const unsigned dbase = sx_base + (unsigned)(((t + 1) & 1) * SXB * 2);
#pragma unroll
      for (int i = 0; i < 4; ++i) {
        unsigned g = (unsigned)(i * 256 + tid) * 16;
        unsigned l = (g >> 10) * (LDSK * 2) + (g & 1023);
        async_copy_b128(dbase + l, g, xsrc);
      }
      wait_async4();            // x_t copies (issued one step ago) have landed
    } else {
      wait_async0();
    }
    __syncthreads();            // all waves' slices of sx[t&1] visible

    // ---- 16x16 += 16x512 * 512x16 : hand-sequenced depth-2 pipeline ----
    v8f acc = {};
    if (part == 0) {
      const unsigned baddr = sx_base + (unsigned)((t & 1) * SXB * 2) + lvoff;
      v8bf lo[3], hi[3];
      ld_lds_pair(lo[0], hi[0], baddr, 0);
      ld_lds_pair(lo[1], hi[1], baddr, 64);
#pragma unroll
      for (int kt = 0; kt < KT; ++kt) {
        if (kt + 2 < KT) ld_lds_pair(lo[(kt + 2) % 3], hi[(kt + 2) % 3], baddr, (kt + 2) * 64);
        v16bf b = BSHUF(lo[kt % 3], hi[kt % 3]);
        if (kt < KT - 2)       wait_ds4(b);
        else if (kt == KT - 2) wait_ds2(b);
        else                   wait_ds0(b);
        acc = __builtin_amdgcn_wmma_f32_16x16x32_bf16(
                  false, afrag[kt], false, b, (short)0, acc, false, false);
      }
    } else {
      const bf16_t* hbase = hT + (size_t)(t & 1) * HB + (size_t)n0 * H;
      v8bf lo[3], hi[3];
      ld_glb_pair(lo[0], hi[0], bvoff, hbase, 0);
      ld_glb_pair(lo[1], hi[1], bvoff, hbase, 64);
#pragma unroll
      for (int kt = 0; kt < KT; ++kt) {
        if (kt + 2 < KT) ld_glb_pair(lo[(kt + 2) % 3], hi[(kt + 2) % 3], bvoff, hbase, (kt + 2) * 64);
        v16bf b = BSHUF(lo[kt % 3], hi[kt % 3]);
        if (kt < KT - 2)       wait_ld4(b);
        else if (kt == KT - 2) wait_ld2(b);
        else                   wait_ld0(b);
        acc = __builtin_amdgcn_wmma_f32_16x16x32_bf16(
                  false, afrag[kt], false, b, (short)0, acc, false, false);
      }
    }

    // ---- gate partial -> LDS (C/D layout: VGPR i -> M=i / M=8+i) ----
    {
      const int rbase = (lane < 16) ? 0 : 8;
      const int col   = lane & 15;
      float* gb = gacc[wv];
#pragma unroll
      for (int i = 0; i < 8; ++i) gb[(rbase + i) * 16 + col] = acc[i];
    }
    __syncthreads();

    // ---- elementwise cell update + transpose x_{t+2} into the ring ----
    {
      const int ei = em * 16 + en;
      float zi = gacc[0][ei] + gacc[1][ei];
      float zf = gacc[2][ei] + gacc[3][ei];
      float zg = gacc[4][ei] + gacc[5][ei];
      float zo = gacc[6][ei] + gacc[7][ei];
      float iv = 1.0f / (1.0f + __expf(-zi));
      float fv = 1.0f / (1.0f + __expf(-zf));
      float gv = tanhf(zg);
      float ov = 1.0f / (1.0f + __expf(-zo));
      c_reg = fv * c_reg + iv * gv;
      float hv = ov * tanhf(c_reg);

      hT[(size_t)((t + 1) & 1) * HB + tcol] = (bf16_t)hv;     // publish h_t (transposed)
      const size_t oi = (size_t)(m0 + em) * B + n0 + en;
      out[(size_t)t * HB + oi] = hv;                          // outputs[t] == h_t

      if (t + 2 < T_STEPS) {                                  // stage x two steps ahead
        float xv = in_seq[(size_t)(t + 2) * HB + oi];
        xT[(size_t)((t + 2) % 3) * HB + tcol] = (bf16_t)xv;
        if (en == 0 && t + 3 < T_STEPS)
          __builtin_prefetch(&in_seq[(size_t)(t + 3) * HB + (size_t)(m0 + em) * B + n0], 0, 1);
      }
      if (t == T_STEPS - 1) {
        out[(size_t)T_STEPS * HB + oi]      = hv;             // final h
        out[(size_t)T_STEPS * HB + HB + oi] = c_reg;          // final c
      }
    }

    bt += NT_M;
    group_barrier(gbar, bt);   // publishes h_t and xT_{t+2} within the column group
  }
}

extern "C" void kernel_launch(void* const* d_in, const int* in_sizes, int n_in,
                              void* d_out, int out_size, void* d_ws, size_t ws_size,
                              hipStream_t stream) {
  const float* in_seq = (const float*)d_in[0];
  const float* Wxi = (const float*)d_in[1];
  const float* Wxf = (const float*)d_in[2];
  const float* Wxo = (const float*)d_in[3];
  const float* Wxg = (const float*)d_in[4];
  const float* Whi = (const float*)d_in[5];
  const float* Whf = (const float*)d_in[6];
  const float* Who = (const float*)d_in[7];
  const float* Whg = (const float*)d_in[8];

  unsigned char*  ws  = (unsigned char*)d_ws;
  unsigned*       bar = (unsigned*)ws;                         // 4 counters, 64B apart
  unsigned short* xT  = (unsigned short*)(ws + 256);           // bf16 [3][B][H]
  unsigned short* hT  = (unsigned short*)(ws + 256 + 3 * HB * 2);

  hipMemsetAsync(ws, 0, 256, stream);                          // reset barriers (capture-safe)
  lstm_persistent_wmma<<<NGRP, 256, 0, stream>>>(
      in_seq, Wxi, Wxf, Wxo, Wxg, Whi, Whf, Who, Whg,
      (float*)d_out, xT, hT, bar);
}
